// MOE_81698867904814
// MI455X (gfx1250) — compile-verified, matrix operand order checked by reading, not run on hardware
//
#include <hip/hip_runtime.h>

#define NTOK 2048      // B*T
#define C_DIM 768
#define H_DIM 3072
#define NE 8
#define TOPK 2
#define CAP 2048       // worst-case tokens per expert
#define KC 32          // WMMA K per step (bf16)
#define MTILE 64       // gathered rows per block
#define NTILE 128      // output columns per block
#define LDSW 20        // dwords per LDS row (16 data + 4 pad) -> conflict-free b128

typedef __attribute__((ext_vector_type(16))) __bf16 v16bf;
typedef __attribute__((ext_vector_type(8)))  float  v8f;

#if defined(__has_builtin)
#  if __has_builtin(__builtin_amdgcn_global_load_async_to_lds_b128)
#    define HAVE_ASYNC 1
#    define HAVE_ASYNC_B128 1
#  elif __has_builtin(__builtin_amdgcn_global_load_async_to_lds_b64)
#    define HAVE_ASYNC 1           // b64 variant proven present on this toolchain
#  endif
#  if __has_builtin(__builtin_amdgcn_s_wait_asynccnt)
#    define WAIT_ASYNC(n) __builtin_amdgcn_s_wait_asynccnt(n)
#  endif
#endif
#if defined(HAVE_ASYNC) && !defined(WAIT_ASYNC)
#  define WAIT_ASYNC(n) asm volatile("s_wait_asynccnt %0" :: "i"(n) : "memory")
#endif

#ifdef HAVE_ASYNC
typedef __attribute__((ext_vector_type(4))) int v4i_t;
typedef __attribute__((ext_vector_type(2))) int v2i_t;
typedef __attribute__((address_space(1))) v4i_t* g_v4i;
typedef __attribute__((address_space(3))) v4i_t* l_v4i;
typedef __attribute__((address_space(1))) v2i_t* g_v2i;
typedef __attribute__((address_space(3))) v2i_t* l_v2i;
// copy 16 bytes global -> LDS through the async (ASYNCcnt) path
__device__ __forceinline__ void async_copy16(const void* gsrc, void* ldst) {
#ifdef HAVE_ASYNC_B128
    __builtin_amdgcn_global_load_async_to_lds_b128((g_v4i)gsrc, (l_v4i)ldst, 0, 0);
#else
    __builtin_amdgcn_global_load_async_to_lds_b64((g_v2i)gsrc, (l_v2i)ldst, 0, 0);
    __builtin_amdgcn_global_load_async_to_lds_b64(
        (g_v2i)((const char*)gsrc + 8), (l_v2i)((char*)ldst + 8), 0, 0);
#endif
}
#endif

union __align__(16) Frag {
    unsigned int d[8];
    uint4 q[2];
    v16bf v;
};

// fp32->bf16 pair, round-half-up: two v_add + one v_perm
__device__ __forceinline__ unsigned int pack_bf2(float lo, float hi) {
    unsigned int ul = __float_as_uint(lo) + 0x8000u;
    unsigned int uh = __float_as_uint(hi) + 0x8000u;
    return __builtin_amdgcn_perm(uh, ul, 0x07060302u);   // {uh.b3,uh.b2,ul.b3,ul.b2}
}
__device__ __forceinline__ unsigned short f2bf(float f) {
    return (unsigned short)((__float_as_uint(f) + 0x8000u) >> 16);
}
// tanh-GELU rewritten as x * sigmoid(2u): one v_exp + one v_rcp
__device__ __forceinline__ float gelu_fast(float x) {
    float u = 1.5957691216057308f * (x + 0.044715f * x * x * x);
    return x * __builtin_amdgcn_rcpf(1.0f + __expf(-u));
}

// ---------------- kernel 0: zero per-expert counters ----------------
__global__ void moe_init(int* __restrict__ counts) {
    if (threadIdx.x < NE) counts[threadIdx.x] = 0;
}

// ---------------- kernel 1: router (one wave32 per token) ----------------
__global__ __launch_bounds__(256)
void moe_router(const float* __restrict__ x, const float* __restrict__ wg,
                int* __restrict__ counts, int* __restrict__ lists,
                float* __restrict__ p_arr) {
    const int wave = threadIdx.x >> 5;
    const int lane = threadIdx.x & 31;
    const int t = blockIdx.x * 8 + wave;
    if (t >= NTOK) return;

    float acc[NE];
#pragma unroll
    for (int e = 0; e < NE; e++) acc[e] = 0.f;
    const float* xr = x + (size_t)t * C_DIM;
    for (int c = lane; c < C_DIM; c += 32) {
        float xv = xr[c];
#pragma unroll
        for (int e = 0; e < NE; e++) acc[e] += xv * wg[c * NE + e];
    }
#pragma unroll
    for (int e = 0; e < NE; e++) {
#pragma unroll
        for (int off = 16; off > 0; off >>= 1)
            acc[e] += __shfl_xor(acc[e], off, 32);
    }
    float mx = acc[0];
#pragma unroll
    for (int e = 1; e < NE; e++) mx = fmaxf(mx, acc[e]);
    float p[NE], s = 0.f;
#pragma unroll
    for (int e = 0; e < NE; e++) { p[e] = __expf(acc[e] - mx); s += p[e]; }
#pragma unroll
    for (int e = 0; e < NE; e++) p[e] /= s;
    int i0 = 0;
#pragma unroll
    for (int e = 1; e < NE; e++) if (p[e] > p[i0]) i0 = e;
    int i1 = (i0 == 0) ? 1 : 0;
#pragma unroll
    for (int e = 0; e < NE; e++) if (e != i0 && p[e] > p[i1]) i1 = e;
    float inv = 1.0f / (p[i0] + p[i1]);

    if (lane == 0) {
        int s0 = atomicAdd(&counts[i0], 1);
        lists[i0 * CAP + s0] = t * TOPK + 0;
        p_arr[t * TOPK + 0] = p[i0] * inv;
        int s1 = atomicAdd(&counts[i1], 1);
        lists[i1 * CAP + s1] = t * TOPK + 1;
        p_arr[t * TOPK + 1] = p[i1] * inv;
    }
}

// ---- fragment loader: two conflict-free b128 LDS reads, ISA 16-bit layout ----
__device__ __forceinline__ void load_frag(Frag& f, const unsigned int* row_base, int half) {
    const unsigned int* p = row_base + half * 4;
    f.q[0] = *(const uint4*)(p);
    f.q[1] = *(const uint4*)(p + 8);
}

// ---------------- kernel 2: grouped GEMM1 (x @ w_fc[e]) + GELU -> bf16 h ----------------
// 256 threads = 8 waves in 2(M) x 4(N) grid; tile M=64, N=128; each wave: 32x32 out
// (2 A-frags x 2 B-frags = 4 WMMA accums). Double-buffered LDS + double-buffered regs.
__global__ __launch_bounds__(256)
void moe_fc1(const float* __restrict__ x, const float* __restrict__ w_fc,
             const int* __restrict__ counts, const int* __restrict__ lists,
             unsigned short* __restrict__ h_buf) {
    const int e = blockIdx.z;
    const int cnt = counts[e];
    const int rt = blockIdx.y;
    if (rt * MTILE >= cnt) return;
    const int h0 = blockIdx.x * NTILE;
    const int tid = threadIdx.x;
    const int lane = tid & 31;
    const int wv = tid >> 5;
    const int m = lane & 15, half = lane >> 4;
    const int wm = wv >> 2, wn = wv & 3;

    __shared__ __align__(16) unsigned int A_lds[2][MTILE * LDSW];
    __shared__ __align__(16) unsigned int B_lds[2][NTILE * LDSW];
    __shared__ int s_assign[MTILE];
    __shared__ int s_tok[MTILE];

    if (tid < MTILE) {
        int i = rt * MTILE + tid;
        int a = (i < cnt) ? lists[e * CAP + i] : -1;
        s_assign[tid] = a;
        s_tok[tid] = (a >= 0) ? (a >> 1) : (lists[e * CAP + rt * MTILE] >> 1);
    }
    __syncthreads();

    // A staging: one row, 4 dwords (8 floats) per thread
    const int arow = tid >> 2;
    const int adc0 = (tid & 3) * 4;          // dword column base
    const float* a_base = x + (size_t)s_tok[arow] * C_DIM + 2 * adc0;
    // B staging: 8 dwords (16 strided floats) per thread
    const int bn = tid & 127;
    const int bdk0 = (tid >> 7) * 8;
    const size_t cs = (size_t)H_DIM * NE;    // c-stride in floats
    const float* b_base = w_fc + ((size_t)(2 * bdk0) * H_DIM + (h0 + bn)) * NE + e;

    float aR0[8], bR0[16], aR1[8], bR1[16];

#define FC1_LOAD(K0, AR, BR) do {                                            \
        const float4 av0 = *(const float4*)(a_base + (K0));                  \
        const float4 av1 = *(const float4*)(a_base + (K0) + 4);              \
        AR[0]=av0.x; AR[1]=av0.y; AR[2]=av0.z; AR[3]=av0.w;                  \
        AR[4]=av1.x; AR[5]=av1.y; AR[6]=av1.z; AR[7]=av1.w;                  \
        const float* pb = b_base + (size_t)(K0) * cs;                        \
        _Pragma("unroll")                                                    \
        for (int j = 0; j < 16; j++) BR[j] = pb[(size_t)j * cs];             \
    } while (0)

#define FC1_STORE(BUF, AR, BR) do {                                          \
        unsigned int* Ab = &A_lds[BUF][arow * LDSW + adc0];                  \
        _Pragma("unroll")                                                    \
        for (int j = 0; j < 4; j++) Ab[j] = pack_bf2(AR[2*j], AR[2*j+1]);    \
        unsigned int* Bb = &B_lds[BUF][bn * LDSW + bdk0];                    \
        _Pragma("unroll")                                                    \
        for (int j = 0; j < 8; j++) Bb[j] = pack_bf2(BR[2*j], BR[2*j+1]);    \
    } while (0)

    v8f acc[2][2] = {};
    const int n0 = wn * 32 + m;

#define FC_COMPUTE(BUF) do {                                                 \
        Frag b0, b1, a0, a1;                                                 \
        load_frag(b0, &B_lds[BUF][n0 * LDSW], half);                         \
        load_frag(b1, &B_lds[BUF][(n0 + 16) * LDSW], half);                  \
        load_frag(a0, &A_lds[BUF][(wm * 32 + m) * LDSW], half);              \
        load_frag(a1, &A_lds[BUF][(wm * 32 + 16 + m) * LDSW], half);         \
        acc[0][0] = __builtin_amdgcn_wmma_f32_16x16x32_bf16(false, a0.v, false, b0.v, (short)0, acc[0][0], false, false); \
        acc[0][1] = __builtin_amdgcn_wmma_f32_16x16x32_bf16(false, a0.v, false, b1.v, (short)0, acc[0][1], false, false); \
        acc[1][0] = __builtin_amdgcn_wmma_f32_16x16x32_bf16(false, a1.v, false, b0.v, (short)0, acc[1][0], false, false); \
        acc[1][1] = __builtin_amdgcn_wmma_f32_16x16x32_bf16(false, a1.v, false, b1.v, (short)0, acc[1][1], false, false); \
    } while (0)

    const int NCH = C_DIM / KC;   // 24 (even)
    FC1_LOAD(0, aR0, bR0);
    for (int kc = 0; kc < NCH; kc += 2) {
        // phase 0: compute buf0 from reg-set 0; prefetch chunk kc+1 into set 1
        FC1_LOAD((kc + 1) * KC, aR1, bR1);
        FC1_STORE(0, aR0, bR0);
        __syncthreads();
        FC_COMPUTE(0);
        // phase 1: compute buf1 from reg-set 1; prefetch chunk kc+2 into set 0
        if (kc + 2 < NCH) FC1_LOAD((kc + 2) * KC, aR0, bR0);
        FC1_STORE(1, aR1, bR1);
        __syncthreads();
        FC_COMPUTE(1);
    }

    // epilogue: GELU, cast bf16, scatter to per-assignment h rows
#pragma unroll
    for (int im = 0; im < 2; im++) {
#pragma unroll
        for (int v = 0; v < 8; v++) {
            int row = wm * 32 + im * 16 + v + half * 8;
            int a = s_assign[row];
            if (a >= 0) {
                size_t base = (size_t)a * H_DIM + h0 + wn * 32 + m;
                h_buf[base]      = f2bf(gelu_fast(acc[im][0][v]));
                h_buf[base + 16] = f2bf(gelu_fast(acc[im][1][v]));
            }
        }
    }
#undef FC1_LOAD
#undef FC1_STORE
#undef FC_COMPUTE
}

// ---------------- kernel 3: grouped GEMM2 (h @ w_proj[e]) * gate -> partials ----------------
__global__ __launch_bounds__(256)
void moe_fc2(const unsigned short* __restrict__ h_buf, const float* __restrict__ w_proj,
             const int* __restrict__ counts, const int* __restrict__ lists,
             const float* __restrict__ p_arr, float* __restrict__ o_part) {
    const int e = blockIdx.z;
    const int cnt = counts[e];
    const int rt = blockIdx.y;
    if (rt * MTILE >= cnt) return;
    const int c0 = blockIdx.x * NTILE;
    const int tid = threadIdx.x;
    const int lane = tid & 31;
    const int wv = tid >> 5;
    const int m = lane & 15, half = lane >> 4;
    const int wm = wv >> 2, wn = wv & 3;

    __shared__ __align__(16) unsigned int A_lds[2][MTILE * LDSW];
    __shared__ __align__(16) unsigned int B_lds[2][NTILE * LDSW];
    __shared__ int s_a[MTILE];
    __shared__ int s_g[MTILE];
    __shared__ float s_p[MTILE];

    if (tid < MTILE) {
        int i = rt * MTILE + tid;
        int a = (i < cnt) ? lists[e * CAP + i] : -1;
        s_a[tid] = a;
        s_g[tid] = (a >= 0) ? a : lists[e * CAP + rt * MTILE];
        s_p[tid] = (a >= 0) ? p_arr[a] : 0.f;
    }
    __syncthreads();

    // A tile: bf16 pass-through, one 16B async copy per thread per chunk
    const int arow = tid >> 2;
    const int adc0 = (tid & 3) * 4;
    const unsigned short* h_src = h_buf + (size_t)s_g[arow] * H_DIM + 2 * adc0;
    const int bn = tid & 127;
    const int bdk0 = (tid >> 7) * 8;
    const size_t ks = (size_t)C_DIM * NE;    // h-stride in floats
    const float* b_base = w_proj + ((size_t)(2 * bdk0) * C_DIM + (c0 + bn)) * NE + e;

    float bR0[16], bR1[16];
#ifndef HAVE_ASYNC
    uint4 aU0, aU1;
#endif

#define FC2_LOADB(K0, BR) do {                                               \
        const float* pb = b_base + (size_t)(K0) * ks;                        \
        _Pragma("unroll")                                                    \
        for (int j = 0; j < 16; j++) BR[j] = pb[(size_t)j * ks];             \
    } while (0)
#define FC2_STOREB(BUF, BR) do {                                             \
        unsigned int* Bb = &B_lds[BUF][bn * LDSW + bdk0];                    \
        _Pragma("unroll")                                                    \
        for (int j = 0; j < 8; j++) Bb[j] = pack_bf2(BR[2*j], BR[2*j+1]);    \
    } while (0)

    v8f acc[2][2] = {};
    const int n0 = wn * 32 + m;

#define FC_COMPUTE(BUF) do {                                                 \
        Frag b0, b1, a0, a1;                                                 \
        load_frag(b0, &B_lds[BUF][n0 * LDSW], half);                         \
        load_frag(b1, &B_lds[BUF][(n0 + 16) * LDSW], half);                  \
        load_frag(a0, &A_lds[BUF][(wm * 32 + m) * LDSW], half);              \
        load_frag(a1, &A_lds[BUF][(wm * 32 + 16 + m) * LDSW], half);         \
        acc[0][0] = __builtin_amdgcn_wmma_f32_16x16x32_bf16(false, a0.v, false, b0.v, (short)0, acc[0][0], false, false); \
        acc[0][1] = __builtin_amdgcn_wmma_f32_16x16x32_bf16(false, a0.v, false, b1.v, (short)0, acc[0][1], false, false); \
        acc[1][0] = __builtin_amdgcn_wmma_f32_16x16x32_bf16(false, a1.v, false, b0.v, (short)0, acc[1][0], false, false); \
        acc[1][1] = __builtin_amdgcn_wmma_f32_16x16x32_bf16(false, a1.v, false, b1.v, (short)0, acc[1][1], false, false); \
    } while (0)

    const int NCH = H_DIM / KC;   // 96 (even)
    unsigned int* A0 = &A_lds[0][arow * LDSW + adc0];
    unsigned int* A1 = &A_lds[1][arow * LDSW + adc0];

    // prologue: chunk 0
#ifdef HAVE_ASYNC
    async_copy16(h_src, A0);
#else
    aU0 = *(const uint4*)(h_src);
#endif
    FC2_LOADB(0, bR0);

    for (int kc = 0; kc < NCH; kc += 2) {
        // ---- phase 0: buf0 / reg-set 0 ----
        FC2_LOADB((kc + 1) * KC, bR1);
        FC2_STOREB(0, bR0);
#ifdef HAVE_ASYNC
        WAIT_ASYNC(0);          // chunk kc's A tile landed in buf0
#else
        aU1 = *(const uint4*)(h_src + (kc + 1) * KC);
        *(uint4*)A0 = aU0;
#endif
        __syncthreads();
        FC_COMPUTE(0);
#ifdef HAVE_ASYNC
        // issue next A tile after the barrier: buf1's previous readers are done
        async_copy16(h_src + (kc + 1) * KC, A1);
#endif
        // ---- phase 1: buf1 / reg-set 1 ----
        if (kc + 2 < NCH) FC2_LOADB((kc + 2) * KC, bR0);
        FC2_STOREB(1, bR1);
#ifdef HAVE_ASYNC
        WAIT_ASYNC(0);
#else
        if (kc + 2 < NCH) aU0 = *(const uint4*)(h_src + (kc + 2) * KC);
        *(uint4*)A1 = aU1;
#endif
        __syncthreads();
        FC_COMPUTE(1);
#ifdef HAVE_ASYNC
        if (kc + 2 < NCH) async_copy16(h_src + (kc + 2) * KC, A0);
#endif
    }

#pragma unroll
    for (int im = 0; im < 2; im++) {
#pragma unroll
        for (int v = 0; v < 8; v++) {
            int row = wm * 32 + im * 16 + v + half * 8;
            int a = s_a[row];
            if (a >= 0) {
                float pv = s_p[row];
                size_t base = (size_t)a * C_DIM + c0 + wn * 32 + m;
                o_part[base]      = pv * acc[im][0][v];
                o_part[base + 16] = pv * acc[im][1][v];
            }
        }
    }
#undef FC2_LOADB
#undef FC2_STOREB
#undef FC_COMPUTE
}

// ---------------- kernel 4: combine the two expert partials per token ----------------
__global__ __launch_bounds__(256)
void moe_combine(const float* __restrict__ o_part, float* __restrict__ out) {
    int i = blockIdx.x * 256 + threadIdx.x;
    if (i >= NTOK * C_DIM) return;
    int t = i / C_DIM, c = i % C_DIM;
    out[i] = o_part[(size_t)(t * 2) * C_DIM + c] + o_part[(size_t)(t * 2 + 1) * C_DIM + c];
}

extern "C" void kernel_launch(void* const* d_in, const int* in_sizes, int n_in,
                              void* d_out, int out_size, void* d_ws, size_t ws_size,
                              hipStream_t stream) {
    (void)in_sizes; (void)n_in; (void)out_size; (void)ws_size;
    const float* x      = (const float*)d_in[0];
    const float* w_fc   = (const float*)d_in[1];
    const float* w_proj = (const float*)d_in[2];
    const float* w_gate = (const float*)d_in[3];
    float* out = (float*)d_out;

    char* ws = (char*)d_ws;
    size_t off = 0;
    int* counts = (int*)(ws + off);              off += 256;
    int* lists  = (int*)(ws + off);              off += (size_t)NE * CAP * sizeof(int);
    float* p_arr = (float*)(ws + off);           off += (size_t)NTOK * TOPK * sizeof(float);
    unsigned short* h_buf = (unsigned short*)(ws + off);
    off += (size_t)NTOK * TOPK * H_DIM * sizeof(unsigned short);
    float* o_part = (float*)(ws + off);

    hipLaunchKernelGGL(moe_init, dim3(1), dim3(32), 0, stream, counts);
    hipLaunchKernelGGL(moe_router, dim3(NTOK / 8), dim3(256), 0, stream,
                       x, w_gate, counts, lists, p_arr);
    hipLaunchKernelGGL(moe_fc1, dim3(H_DIM / NTILE, CAP / MTILE, NE), dim3(256), 0, stream,
                       x, w_fc, counts, lists, h_buf);
    hipLaunchKernelGGL(moe_fc2, dim3(C_DIM / NTILE, CAP / MTILE, NE), dim3(256), 0, stream,
                       h_buf, w_proj, counts, lists, p_arr, o_part);
    hipLaunchKernelGGL(moe_combine, dim3((NTOK * C_DIM + 255) / 256), dim3(256), 0, stream,
                       o_part, out);
}
